// SparseGroupingLayer_80470507258197
// MI455X (gfx1250) — matrix operand order repaired and checked by reading, not compiled, not run on hardware
//
#include <hip/hip_runtime.h>
#include <hip/hip_bf16.h>
#include <cmath>

typedef _Float16 h8   __attribute__((ext_vector_type(8)));
typedef _Float16 v16h __attribute__((ext_vector_type(16)));
typedef float    v8f  __attribute__((ext_vector_type(8)));
typedef unsigned int u32x4 __attribute__((ext_vector_type(4)));
typedef int          i32x8 __attribute__((ext_vector_type(8)));
typedef int          i32x4 __attribute__((ext_vector_type(4)));

#define B_    8
#define HI_   64
#define WI_   64
#define D_    128
#define HO_   32
#define WO_   32
#define NIN_  4096
#define NOUT_ 1024
#define EPSF  1e-6f

// ---------------------------------------------------------------- utilities

__device__ __forceinline__ float wred(float v) {
  #pragma unroll
  for (int m = 16; m; m >>= 1) v += __shfl_xor(v, m, 32);
  return v;
}

// CDNA5 16-bit A-fragment (16x32, MxK): lane&15 = row M; lanes<16 hold
// K = [k0..k0+7],[k0+16..k0+23]; lanes>=16 hold K = [k0+8..k0+15],[k0+24..k0+31].
__device__ __forceinline__ v16h load_a_frag(const _Float16* A, int lda, int row,
                                            int k0, int lane) {
  int koff = (lane >> 4) << 3;
  const _Float16* p = A + (long)row * lda + k0 + koff;
  h8 lo = *(const h8*)p;
  h8 hi = *(const h8*)(p + 16);
  v16h a;
  #pragma unroll
  for (int i = 0; i < 8; ++i) { a[i] = lo[i]; a[i + 8] = hi[i]; }
  return a;
}

// CDNA5 16-bit B-fragment (32x16, KxN) for C = A * W^T with W stored N x K
// row-major: lane&15 = column N; lanes<16 hold K=[k0..k0+7],[k0+16..k0+23],
// lanes>=16 hold K=[k0+8..k0+15],[k0+24..k0+31] (same as A-fragment).
__device__ __forceinline__ v16h load_b_frag(const _Float16* W, int ldw, int col,
                                            int k0, int lane) {
  int kb = k0 + ((lane >> 4) << 3);
  const _Float16* p = W + (long)col * ldw + kb;
  h8 lo = *(const h8*)p;
  h8 hi = *(const h8*)(p + 16);
  v16h b;
  #pragma unroll
  for (int i = 0; i < 8; ++i) { b[i] = lo[i]; b[i + 8] = hi[i]; }
  return b;
}

// ----------------------------------------------------- Tensor Data Mover D#
// DMA a [rows x rowlen] f16 tile (row stride = `stride` elements) from global
// memory into LDS at byte offset ldsOff. 2-D tensor: descriptor groups 2/3
// (and the trailing wide group slot of the 6-arg builtin) are zero.
// Group layouts per CDNA5 ISA ch.8.
__device__ __forceinline__ void tdm_load_tile_f16(const _Float16* gsrc,
                                                  unsigned ldsOff, int rows,
                                                  int rowlen, int stride) {
  unsigned long long ga = (unsigned long long)(uintptr_t)gsrc;
  u32x4 g0;
  g0[0] = 1u;                                     // count=1 (user descriptor)
  g0[1] = ldsOff;                                 // lds_addr (bytes)
  g0[2] = (unsigned)(ga & 0xffffffffu);           // global_addr[31:0]
  g0[3] = (unsigned)((ga >> 32) & 0x1ffffffu)     // global_addr[56:32]
          | (2u << 30);                           // type=2 ("image")
  unsigned td0 = (unsigned)rowlen;                // tensor_dim0 (elements)
  unsigned td1 = (unsigned)rows;                  // tensor_dim1
  unsigned long long s0 = (unsigned long long)stride;  // tensor_dim0_stride
  i32x8 g1;
  g1[0] = (int)(1u << 16);                        // data_size=1 -> 2 bytes
  g1[1] = (int)((td0 & 0xffffu) << 16);           // [63:48] = tensor_dim0 lo
  g1[2] = (int)((td0 >> 16) | ((td1 & 0xffffu) << 16));
  g1[3] = (int)((td1 >> 16) | ((td0 & 0xffffu) << 16));  // tile_dim0 = rowlen
  g1[4] = (int)(td1 & 0xffffu);                   // tile_dim1 = rows
  g1[5] = (int)(s0 & 0xffffffffu);                // dim0_stride lo32
  g1[6] = (int)((s0 >> 32) & 0xffffu);            // dim0_stride hi16
  g1[7] = 0;
  i32x4 z4 = {0, 0, 0, 0};
  i32x8 z8 = {0, 0, 0, 0, 0, 0, 0, 0};
  __builtin_amdgcn_tensor_load_to_lds(g0, g1, z4, z4, z8, 0);
}

// ---------------------------------------------------------------- weight cvt

__global__ void convert_weights(const float* wq, const float* wk, const float* wv,
                                const float* w1, const float* w2, const float* sw,
                                _Float16* wq16, _Float16* wk16, _Float16* wv16,
                                _Float16* w116, _Float16* w216, _Float16* sw16) {
  int id = blockIdx.x * blockDim.x + threadIdx.x;
  if (id < 16384)        wq16[id]           = (_Float16)wq[id];
  else if (id < 32768)   wk16[id - 16384]   = (_Float16)wk[id - 16384];
  else if (id < 49152)   wv16[id - 32768]   = (_Float16)wv[id - 32768];
  else if (id < 81920)   w116[id - 49152]   = (_Float16)w1[id - 49152];
  else if (id < 114688)  w216[id - 81920]   = (_Float16)w2[id - 81920];
  else if (id < 262144) {
    // seed_w (O,I,3,3) -> 9 matrices sw16[j][o][i], j = ky*3+kx
    int dd = id - 114688;
    int j = dd / 16384, oc = dd - j * 16384;
    sw16[(long)j * 16384 + oc] = (_Float16)sw[(long)oc * 9 + j];
  }
}

// ---------------------------------------------------------------- layernorms

__global__ void ln_in_kernel(const float* x, const float* g, const float* bt,
                             _Float16* lnx16, _Float16* x16) {
  int wave = threadIdx.x >> 5, lane = threadIdx.x & 31;
  long row = (long)blockIdx.x * 8 + wave;
  const float* p = x + row * D_ + lane * 4;
  float4 v = *(const float4*)p;
  float mu = wred(v.x + v.y + v.z + v.w) * (1.f / D_);
  float d0 = v.x - mu, d1 = v.y - mu, d2 = v.z - mu, d3 = v.w - mu;
  float var = wred(d0 * d0 + d1 * d1 + d2 * d2 + d3 * d3) * (1.f / D_);
  float r = rsqrtf(var + 1e-5f);
  int c = lane * 4;
  _Float16* q = lnx16 + row * D_ + c;
  q[0] = (_Float16)(d0 * r * g[c + 0] + bt[c + 0]);
  q[1] = (_Float16)(d1 * r * g[c + 1] + bt[c + 1]);
  q[2] = (_Float16)(d2 * r * g[c + 2] + bt[c + 2]);
  q[3] = (_Float16)(d3 * r * g[c + 3] + bt[c + 3]);
  _Float16* xq = x16 + row * D_ + c;
  xq[0] = (_Float16)v.x; xq[1] = (_Float16)v.y;
  xq[2] = (_Float16)v.z; xq[3] = (_Float16)v.w;
}

__global__ void ln_f32_kernel(const float* src, const float* g, const float* bt,
                              float* dst) {
  int wave = threadIdx.x >> 5, lane = threadIdx.x & 31;
  long row = (long)blockIdx.x * 8 + wave;
  const float* p = src + row * D_ + lane * 4;
  float4 v = *(const float4*)p;
  float mu = wred(v.x + v.y + v.z + v.w) * (1.f / D_);
  float d0 = v.x - mu, d1 = v.y - mu, d2 = v.z - mu, d3 = v.w - mu;
  float var = wred(d0 * d0 + d1 * d1 + d2 * d2 + d3 * d3) * (1.f / D_);
  float r = rsqrtf(var + 1e-5f);
  int c = lane * 4;
  float4 o;
  o.x = d0 * r * g[c + 0] + bt[c + 0];
  o.y = d1 * r * g[c + 1] + bt[c + 1];
  o.z = d2 * r * g[c + 2] + bt[c + 2];
  o.w = d3 * r * g[c + 3] + bt[c + 3];
  *(float4*)(dst + row * D_ + c) = o;
}

__global__ void ln_f16_kernel(const float* src, const float* g, const float* bt,
                              _Float16* dst) {
  int wave = threadIdx.x >> 5, lane = threadIdx.x & 31;
  long row = (long)blockIdx.x * 8 + wave;
  const float* p = src + row * D_ + lane * 4;
  float4 v = *(const float4*)p;
  float mu = wred(v.x + v.y + v.z + v.w) * (1.f / D_);
  float d0 = v.x - mu, d1 = v.y - mu, d2 = v.z - mu, d3 = v.w - mu;
  float var = wred(d0 * d0 + d1 * d1 + d2 * d2 + d3 * d3) * (1.f / D_);
  float r = rsqrtf(var + 1e-5f);
  int c = lane * 4;
  _Float16* q = dst + row * D_ + c;
  q[0] = (_Float16)(d0 * r * g[c + 0] + bt[c + 0]);
  q[1] = (_Float16)(d1 * r * g[c + 1] + bt[c + 1]);
  q[2] = (_Float16)(d2 * r * g[c + 2] + bt[c + 2]);
  q[3] = (_Float16)(d3 * r * g[c + 3] + bt[c + 3]);
}

__global__ void add_ln_kernel(const float* t, const float* g, const float* bt,
                              float* xout) {
  int wave = threadIdx.x >> 5, lane = threadIdx.x & 31;
  long row = (long)blockIdx.x * 8 + wave;
  const float* p = t + row * D_ + lane * 4;
  float4 v = *(const float4*)p;
  float mu = wred(v.x + v.y + v.z + v.w) * (1.f / D_);
  float d0 = v.x - mu, d1 = v.y - mu, d2 = v.z - mu, d3 = v.w - mu;
  float var = wred(d0 * d0 + d1 * d1 + d2 * d2 + d3 * d3) * (1.f / D_);
  float r = rsqrtf(var + 1e-5f);
  int c = lane * 4;
  float* o = xout + row * D_ + c;
  o[0] += d0 * r * g[c + 0] + bt[c + 0];
  o[1] += d1 * r * g[c + 1] + bt[c + 1];
  o[2] += d2 * r * g[c + 2] + bt[c + 2];
  o[3] += d3 * r * g[c + 3] + bt[c + 3];
}

// ---------------------------------------------------------------- WMMA GEMM
// C(MxN) = A(MxK,f16) * W(NxK,f16)^T. A-tile (16 x K) is DMA'd to LDS by the
// Tensor Data Mover; all 8 waves read A-fragments via ds_load, B-fragments
// stream from L2. EPI: 0 f32, 1 f16, 2 f16+unfold remap, 3 bias+GELU->f16,
// 4 bias->f32.

template <int K, int EPI>
__global__ void gemm_wmma(const _Float16* __restrict__ A,
                          const _Float16* __restrict__ W,
                          const float* __restrict__ bias,
                          float* Cf, _Float16* Ch, int N) {
  __shared__ __align__(16) _Float16 atile[16 * K];
  int lane = threadIdx.x & 31;
  int wave = threadIdx.x >> 5;
  int m0 = blockIdx.x << 4;
  int n0 = (blockIdx.y << 7) + (wave << 4);
  if (threadIdx.x < 32) {  // wave 0 issues the TDM load, waits on TENSORcnt
    tdm_load_tile_f16(A + (long)m0 * K, (unsigned)(uintptr_t)atile, 16, K, K);
    __builtin_amdgcn_s_wait_tensorcnt(0);
  }
  __syncthreads();
  int bcol = n0 + (lane & 15);
  v8f c = {};
  #pragma unroll
  for (int k0 = 0; k0 < K; k0 += 32) {
    v16h a = load_a_frag(atile, K, lane & 15, k0, lane);
    v16h b = load_b_frag(W, K, bcol, k0, lane);
    c = __builtin_amdgcn_wmma_f32_16x16x32_f16(false, a, false, b,
                                               (short)0, c, false, false);
  }
  int col = n0 + (lane & 15);
  int rbase = (lane >> 4) << 3;  // C/D layout: VGPR r -> M = r (lanes<16) / r+8
  float bv = (EPI >= 3) ? bias[col] : 0.f;
  #pragma unroll
  for (int r = 0; r < 8; ++r) {
    int row = m0 + rbase + r;
    float v = c[r];
    if (EPI == 0) {
      Cf[(long)row * N + col] = v;
    } else if (EPI == 1) {
      Ch[(long)row * N + col] = (_Float16)v;
    } else if (EPI == 2) {
      // row = b*4096 + hi*64 + wi  ->  [(b*4+head)*1024 + oy*32+ox]*128 + col
      int bb = row >> 12, rem = row & 4095;
      int hi = rem >> 6, wi = rem & 63;
      int hd = ((hi & 1) << 1) | (wi & 1);
      int oy = hi >> 1, ox = wi >> 1;
      long o = ((((long)bb * 4 + hd) * NOUT_) + oy * WO_ + ox) * D_ + col;
      Ch[o] = (_Float16)v;
    } else if (EPI == 3) {
      float t = v + bv;
      float gl = 0.5f * t * (1.f + erff(t * 0.70710678118654752f));
      Ch[(long)row * N + col] = (_Float16)gl;
    } else {  // EPI == 4
      Cf[(long)row * N + col] = v + bv;
    }
  }
}

// ------------------------------------------------ seed conv as 9 WMMA GEMMs
// seed[b, oy*32+ox, o] = sum_j W_j[o,:] . x[b, 2oy-1+ky, 2ox-1+kx, :]
// OOB columns read from a zeroed scratch page (pointer select, no per-element
// cndmask).
__global__ void conv_seed_wmma(const _Float16* __restrict__ x16,
                               const _Float16* __restrict__ sw16,
                               const _Float16* __restrict__ zpad,
                               float* __restrict__ seed) {
  int lane = threadIdx.x & 31;
  int wave = threadIdx.x >> 5;
  int m0 = blockIdx.x << 4;                 // 16 output positions (same b, oy)
  int bb = m0 >> 10, rem = m0 & 1023;
  int oy = rem >> 5, ox0 = rem & 31;        // ox0 in {0,16}
  int n0 = wave << 4;
  int lrow = lane & 15;
  int ox = ox0 + lrow;
  int koff = (lane >> 4) << 3;
  v8f c = {};
  for (int ky = 0; ky < 3; ++ky) {
    int iy = 2 * oy - 1 + ky;
    if (iy < 0 || iy >= HI_) continue;      // wave-uniform
    #pragma unroll
    for (int kx = 0; kx < 3; ++kx) {
      int ix = 2 * ox - 1 + kx;
      bool ok = (ix >= 0) && (ix < WI_);
      const _Float16* Ap =
          ok ? (x16 + ((((long)bb * HI_ + iy) * WI_) + ix) * D_) : zpad;
      const _Float16* Wj = sw16 + (ky * 3 + kx) * 16384;
      #pragma unroll
      for (int k0 = 0; k0 < D_; k0 += 32) {
        h8 lo = *(const h8*)(Ap + k0 + koff);
        h8 hi = *(const h8*)(Ap + k0 + koff + 16);
        v16h a;
        #pragma unroll
        for (int i = 0; i < 8; ++i) { a[i] = lo[i]; a[i + 8] = hi[i]; }
        v16h b = load_b_frag(Wj, D_, n0 + lrow, k0, lane);
        c = __builtin_amdgcn_wmma_f32_16x16x32_f16(false, a, false, b,
                                                   (short)0, c, false, false);
      }
    }
  }
  int col = n0 + (lane & 15);
  int rbase = (lane >> 4) << 3;
  #pragma unroll
  for (int r = 0; r < 8; ++r) seed[(long)(m0 + rbase + r) * D_ + col] = c[r];
}

// ------------------------------------------------------------- attention

__global__ void attn_kernel(const _Float16* __restrict__ q16,
                            const _Float16* __restrict__ k16,
                            const float* __restrict__ rpb,
                            const float* __restrict__ tau,
                            float* __restrict__ attn9) {
  int t = blockIdx.x * blockDim.x + threadIdx.x;
  if (t >= B_ * 4 * NOUT_) return;
  int b = t >> 12, rem = t & 4095;
  int h = rem >> 10, i = rem & 1023;
  int oy = i >> 5, ox = i & 31;
  int cy = oy - 1; cy = cy < 0 ? 0 : (cy > HO_ - 3 ? HO_ - 3 : cy);
  int cx = ox - 1; cx = cx < 0 ? 0 : (cx > WO_ - 3 ? WO_ - 3 : cx);
  const _Float16* qp = q16 + ((long)b * NOUT_ + i) * D_;
  float scale = expf(tau[0]);
  float l[9], mx = -3.4e38f;
  #pragma unroll
  for (int ky = 0; ky < 3; ++ky)
    for (int kx = 0; kx < 3; ++kx) {
      const _Float16* kp =
          k16 + ((((long)b * 4 + h) * NOUT_) + (cy + ky) * WO_ + (cx + kx)) * D_;
      float d = 0.f;
      #pragma unroll 8
      for (int c = 0; c < D_; ++c) d += (float)qp[c] * (float)kp[c];
      float lg = scale * (d + rpb[h * 9 + ky * 3 + kx]);
      l[ky * 3 + kx] = lg;
      mx = fmaxf(mx, lg);
    }
  float s = 0.f;
  #pragma unroll
  for (int j = 0; j < 9; ++j) { l[j] = expf(l[j] - mx); s += l[j]; }
  float inv = 1.f / s;
  float* ap = attn9 + (long)t * 9;
  #pragma unroll
  for (int j = 0; j < 9; ++j) ap[j] = l[j] * inv + EPSF;
}

__global__ void update_kernel(const float* __restrict__ attn9,
                              const _Float16* __restrict__ v16,
                              float* __restrict__ xout,
                              _Float16* __restrict__ xo16,
                              float* __restrict__ den) {
  int bi = blockIdx.x;             // b*1024 + i
  int b = bi >> 10, i = bi & 1023;
  int oy = i >> 5, ox = i & 31;
  __shared__ float wsh[36];
  __shared__ float part[4];
  __shared__ float sden;
  int tid = threadIdx.x;
  if (tid < 4) {
    int h = tid;
    long base = ((long)(b * 4 + h) * NOUT_ + i) * 9;
    float a00 = attn9[((long)(b * 4 + h) * NOUT_) * 9];  // attn[b,h,0,0]
    float s = 0.f; int nInv = 0;
    #pragma unroll
    for (int t9 = 0; t9 < 9; ++t9) {
      int ky = t9 / 3, kx = t9 - 3 * (t9 / 3);
      int ny = oy - 1 + ky, nx = ox - 1 + kx;
      bool val = (ny >= 0) && (ny < HO_) && (nx >= 0) && (nx < WO_);
      float av = attn9[base + t9];
      if (val) s += av; else ++nInv;
      wsh[h * 9 + t9] = val ? av : a00;   // attn36 value (columns 0..8)
    }
    part[h] = 4.f * (s + (float)nInv * a00);
  }
  __syncthreads();
  if (tid == 0) {
    sden = part[0] + part[1] + part[2] + part[3] + EPSF;
    den[bi] = sden;
  }
  __syncthreads();
  if (tid < 36) wsh[tid] /= sden;          // a_col_h[..., :9]
  __syncthreads();
  int cy = oy - 1; cy = cy < 0 ? 0 : (cy > HO_ - 3 ? HO_ - 3 : cy);
  int cx = ox - 1; cx = cx < 0 ? 0 : (cx > WO_ - 3 ? WO_ - 3 : cx);
  int d = tid;
  float acc = 0.f;
  #pragma unroll
  for (int h = 0; h < 4; ++h)
    for (int ky = 0; ky < 3; ++ky)
      for (int kx = 0; kx < 3; ++kx)
        acc += wsh[h * 9 + ky * 3 + kx] *
               (float)v16[((((long)b * 4 + h) * NOUT_) +
                           (cy + ky) * WO_ + (cx + kx)) * D_ + d];
  long o = ((long)b * NOUT_ + i) * D_ + d;
  float nv = xout[o] + acc;
  xout[o] = nv;
  xo16[o] = (_Float16)nv;
}

__global__ void scatter_kernel(const float* __restrict__ attn9,
                               const float* __restrict__ den,
                               float* __restrict__ A_row,
                               float* __restrict__ A_col) {
  int t = blockIdx.x * blockDim.x + threadIdx.x;
  if (t >= B_ * 4 * NOUT_) return;
  int b = t >> 12, rem = t & 4095;
  int h = rem >> 10, i = rem & 1023;
  int oy = i >> 5, ox = i & 31;
  const float* ap = attn9 + (long)t * 9;
  float a00 = attn9[(long)(t - i) * 9];
  float D = den[b * NOUT_ + i];
  float* Arow = A_row + ((long)b * NIN_ + h * NOUT_ + i) * NOUT_;
  #pragma unroll
  for (int t9 = 0; t9 < 9; ++t9) {
    int ky = t9 / 3, kx = t9 - 3 * (t9 / 3);
    int ny = oy - 1 + ky, nx = ox - 1 + kx;
    bool val = (ny >= 0) && (ny < HO_) && (nx >= 0) && (nx < WO_);
    float v = val ? ap[t9] : a00;
    int coarse = val ? (ny * WO_ + nx) : 0;
    atomicAdd(Arow + coarse, 4.f * v);           // 4 identical column repeats
    int rowIn = val ? (coarse * 4 + h) : 0;
    float* Acol = A_col + ((long)b * NIN_ + rowIn) * NOUT_;
    float vn = v / D;
    #pragma unroll
    for (int r = 0; r < 4; ++r) atomicAdd(Acol + (9 * r + t9), vn);
  }
}

// ---------------------------------------------------------------- launcher

extern "C" void kernel_launch(void* const* d_in, const int* in_sizes, int n_in,
                              void* d_out, int out_size, void* d_ws, size_t ws_size,
                              hipStream_t stream) {
  const float* x    = (const float*)d_in[0];
  const float* sw   = (const float*)d_in[1];
  const float* wq   = (const float*)d_in[2];
  const float* wk   = (const float*)d_in[3];
  const float* wv   = (const float*)d_in[4];
  const float* w1   = (const float*)d_in[5];
  const float* b1   = (const float*)d_in[6];
  const float* w2   = (const float*)d_in[7];
  const float* b2   = (const float*)d_in[8];
  const float* lnig = (const float*)d_in[9];
  const float* lnib = (const float*)d_in[10];
  const float* lnog = (const float*)d_in[11];
  const float* lnob = (const float*)d_in[12];
  const float* rpb  = (const float*)d_in[13];
  const float* tau  = (const float*)d_in[14];

  float* xout  = (float*)d_out;                       // (B,1024,128)
  float* A_row = xout + (size_t)B_ * NOUT_ * D_;      // (B,4096,1024)
  float* A_col = A_row + (size_t)B_ * NIN_ * NOUT_;   // (B,4096,1024)

  // ---- workspace carve (f16 region then f32 region) ----
  _Float16* f = (_Float16*)d_ws;
  _Float16* zpad  = f; f += 128;                      // 256B zero page
  _Float16* wq16  = f; f += 16384;
  _Float16* wk16  = f; f += 16384;
  _Float16* wv16  = f; f += 16384;
  _Float16* w116  = f; f += 32768;
  _Float16* w216  = f; f += 32768;
  _Float16* sw16  = f; f += 147456;
  _Float16* lnx16 = f; f += (size_t)B_ * NIN_ * D_;   // LN_in(x)
  _Float16* x16   = f; f += (size_t)B_ * NIN_ * D_;   // x in f16
  _Float16* k16   = f; f += (size_t)B_ * 4 * NOUT_ * D_;  // unfolded k
  _Float16* v16   = f; f += (size_t)B_ * 4 * NOUT_ * D_;  // unfolded v
  _Float16* lnq16 = f; f += (size_t)B_ * NOUT_ * D_;  // LN(x_out)
  _Float16* q16   = f; f += (size_t)B_ * NOUT_ * D_;
  _Float16* xo16  = f; f += (size_t)B_ * NOUT_ * D_;  // post-update x_out f16
  _Float16* h116  = f; f += (size_t)B_ * NOUT_ * 256; // gelu(h1)
  float* fp    = (float*)f;
  float* seed  = fp; fp += (size_t)B_ * NOUT_ * D_;
  float* tbuf  = fp; fp += (size_t)B_ * NOUT_ * D_;
  float* attn9 = fp; fp += (size_t)B_ * 4 * NOUT_ * 9;
  float* den   = fp; fp += (size_t)B_ * NOUT_;

  // zero the sparse outputs + zero page (re-done every call; capturable)
  (void)hipMemsetAsync(A_row, 0, (size_t)2 * B_ * NIN_ * NOUT_ * sizeof(float),
                       stream);
  (void)hipMemsetAsync(zpad, 0, 128 * sizeof(_Float16), stream);

  convert_weights<<<1024, 256, 0, stream>>>(wq, wk, wv, w1, w2, sw,
                                            wq16, wk16, wv16, w116, w216, sw16);

  // LN_in + f16 casts
  ln_in_kernel<<<4096, 256, 0, stream>>>(x, lnig, lnib, lnx16, x16);

  // k = LN_in(x) @ wk^T  (unfold epilogue);  v = x @ wv^T
  gemm_wmma<128, 2><<<dim3(2048, 1), 256, 0, stream>>>(lnx16, wk16, nullptr,
                                                       nullptr, k16, 128);
  gemm_wmma<128, 2><<<dim3(2048, 1), 256, 0, stream>>>(x16, wv16, nullptr,
                                                       nullptr, v16, 128);

  // seed conv (9 accumulated WMMA GEMMs) then x_out = LN(seed)
  conv_seed_wmma<<<512, 256, 0, stream>>>(x16, sw16, zpad, seed);
  ln_f32_kernel<<<1024, 256, 0, stream>>>(seed, lnog, lnob, xout);

  for (int it = 0; it < 3; ++it) {
    // q = LN(x_out) @ wq^T
    ln_f16_kernel<<<1024, 256, 0, stream>>>(xout, lnog, lnob, lnq16);
    gemm_wmma<128, 1><<<dim3(512, 1), 256, 0, stream>>>(lnq16, wq16, nullptr,
                                                        nullptr, q16, 128);
    // neighborhood attention (clamped windows) + softmax + EPS
    attn_kernel<<<128, 256, 0, stream>>>(q16, k16, rpb, tau, attn9);
    // cross-head normalize + v aggregation; x_out += updates
    update_kernel<<<B_ * NOUT_, 128, 0, stream>>>(attn9, v16, xout, xo16, den);
    if (it == 2)  // attn36 / a_col_h of the final iteration feed A_row/A_col
      scatter_kernel<<<128, 256, 0, stream>>>(attn9, den, A_row, A_col);
    // MLP: h1 = gelu(x_out@w1^T+b1); x_out += LN(h1@w2^T+b2)
    gemm_wmma<128, 3><<<dim3(512, 2), 256, 0, stream>>>(xo16, w116, b1,
                                                        nullptr, h116, 256);
    gemm_wmma<256, 4><<<dim3(512, 1), 256, 0, stream>>>(h116, w216, b2,
                                                        tbuf, nullptr, 128);
    add_ln_kernel<<<1024, 256, 0, stream>>>(tbuf, lnog, lnob, xout);
  }
}
